// EfficientASAMLayer_12120397709559
// MI455X (gfx1250) — matrix operand order repaired and checked by reading, not compiled, not run on hardware
//
#include <hip/hip_runtime.h>

// ---------------------------------------------------------------------------
// EfficientASAMLayer for MI455X (gfx1250): bf16 WMMA GEMMs with async-to-LDS
// double-buffered staging + VALU windowed attention. Wave32 everywhere.
// ---------------------------------------------------------------------------

typedef __attribute__((ext_vector_type(16))) __bf16 v16bf;
typedef __attribute__((ext_vector_type(8)))  __bf16 bf16x8;
typedef __attribute__((ext_vector_type(8)))  float  v8f;
typedef __attribute__((ext_vector_type(4)))  int    v4i;

#define S_LEN   2048
#define DMODEL  1024
#define NHEAD   16
#define HDIM    64
#define WHALF   64      // WINDOW // 2
#define LDSPAD  40      // 32 bf16 + 8 pad -> 80B row stride, ds_read_b128 conflict-free

#if __has_builtin(__builtin_amdgcn_global_load_async_to_lds_b128) && \
    __has_builtin(__builtin_amdgcn_s_wait_asynccnt)
#define HAS_ASYNC 1
#else
#define HAS_ASYNC 0
#endif

// Builtin takes v4i* pointers (per compiler diagnostic). Flat->AS casts via
// integer: global flat addr == AS1 addr; low 32 bits of flat LDS addr == AS3.
#define GPTR(p) ((__attribute__((address_space(1))) v4i*)(unsigned long long)(const void*)(p))
#define LPTR(p) ((__attribute__((address_space(3))) v4i*)(unsigned int)(unsigned long long)(void*)(p))

__device__ __forceinline__ unsigned short f2bf(float f) {
  unsigned int u = __float_as_uint(f);
  u += 0x7FFFu + ((u >> 16) & 1u);       // round-to-nearest-even
  return (unsigned short)(u >> 16);
}

// ---------------------------------------------------------------------------
// fp32 -> bf16 weight conversion (grid-stride)
// ---------------------------------------------------------------------------
__global__ void cvt_bf16_kernel(const float* __restrict__ in,
                                unsigned short* __restrict__ out, int n) {
  int i = blockIdx.x * blockDim.x + threadIdx.x;
  int stride = gridDim.x * blockDim.x;
  for (; i < n; i += stride) out[i] = f2bf(in[i]);
}

// ---------------------------------------------------------------------------
// LayerNorm: one row (1024 elems) per 256-thread block; writes bf16 xn
// ---------------------------------------------------------------------------
__global__ void layernorm_kernel(const float* __restrict__ x,
                                 const float* __restrict__ g,
                                 const float* __restrict__ b,
                                 unsigned short* __restrict__ xnb) {
  __shared__ float ssum[256];
  __shared__ float ssq[256];
  const int row = blockIdx.x;
  const int t   = threadIdx.x;
  const float* xr = x + (size_t)row * DMODEL;
  float4 v = ((const float4*)xr)[t];                 // 256 * 4 = 1024
  float s = v.x + v.y + v.z + v.w;
  float q = v.x * v.x + v.y * v.y + v.z * v.z + v.w * v.w;
  ssum[t] = s; ssq[t] = q;
  __syncthreads();
  for (int o = 128; o > 0; o >>= 1) {
    if (t < o) { ssum[t] += ssum[t + o]; ssq[t] += ssq[t + o]; }
    __syncthreads();
  }
  const float mu   = ssum[0] * (1.0f / DMODEL);
  const float var  = ssq[0] * (1.0f / DMODEL) - mu * mu;   // biased var (jnp.var)
  const float rstd = rsqrtf(var + 1e-5f);
  float4 gg = ((const float4*)g)[t];
  float4 bb = ((const float4*)b)[t];
  unsigned short* o = xnb + (size_t)row * DMODEL + t * 4;
  o[0] = f2bf((v.x - mu) * rstd * gg.x + bb.x);
  o[1] = f2bf((v.y - mu) * rstd * gg.y + bb.y);
  o[2] = f2bf((v.z - mu) * rstd * gg.z + bb.z);
  o[3] = f2bf((v.w - mu) * rstd * gg.w + bb.w);
}

// ---------------------------------------------------------------------------
// Stage one 64x32 A tile + 128x32 B tile (bf16) into LDS. 128 threads, 16B
// chunks: A = 256 chunks (2/thread), B = 512 chunks (4/thread), coalesced.
// ---------------------------------------------------------------------------
__device__ __forceinline__ void stage_tiles(const unsigned short* __restrict__ A,
                                            const unsigned short* __restrict__ B,
                                            int K, int m0, int n0, int k0,
                                            unsigned short (*aL)[LDSPAD],
                                            unsigned short (*bL)[LDSPAD],
                                            int tid) {
#pragma unroll
  for (int i = 0; i < 2; ++i) {
    const int c = i * 128 + tid;             // 0..255
    const int row = c >> 2, cc = c & 3;
    const unsigned short* g = A + (size_t)(m0 + row) * K + k0 + cc * 8;
    unsigned short* l = &aL[row][cc * 8];
#if HAS_ASYNC
    __builtin_amdgcn_global_load_async_to_lds_b128(GPTR(g), LPTR(l), 0, 0);
#else
    *(uint4*)l = *(const uint4*)g;
#endif
  }
#pragma unroll
  for (int i = 0; i < 4; ++i) {
    const int c = i * 128 + tid;             // 0..511
    const int row = c >> 2, cc = c & 3;
    const unsigned short* g = B + (size_t)(n0 + row) * K + k0 + cc * 8;
    unsigned short* l = &bL[row][cc * 8];
#if HAS_ASYNC
    __builtin_amdgcn_global_load_async_to_lds_b128(GPTR(g), LPTR(l), 0, 0);
#else
    *(uint4*)l = *(const uint4*)g;
#endif
  }
}

// ---------------------------------------------------------------------------
// GEMM: C[M,N] = A[M,K](bf16) @ B[N,K](bf16)^T  (torch Linear layout)
// Block: 128 threads = 4 waves (2M x 2N); block tile 64x128; wave tile 32x64.
// K-loop: async-to-LDS double buffering; 8 WMMAs + 12 ds_read_b128 per tile.
// Epilogue: +bias, optional exact GELU, +residual; writes fp32 and/or bf16.
// ---------------------------------------------------------------------------
__global__ void __launch_bounds__(128)
gemm_bf16_kernel(const unsigned short* __restrict__ A,
                 const unsigned short* __restrict__ B,
                 int N, int K,
                 const float* __restrict__ bias,
                 const float* __restrict__ res,
                 float* __restrict__ outf,
                 unsigned short* __restrict__ outb,
                 int op /* 0=none, 1=gelu */) {
  __shared__ unsigned short aL[2][64][LDSPAD];     // 2 x 5.0 KB
  __shared__ unsigned short bL[2][128][LDSPAD];    // 2 x 10.0 KB
  const int tid  = threadIdx.x;
  const int lane = tid & 31;
  const int wave = tid >> 5;
  const int l15  = lane & 15;
  const int lh   = lane >> 4;                      // half-wave id
  const int m0 = blockIdx.y * 64;
  const int n0 = blockIdx.x * 128;
  const int m0w = (wave >> 1) * 32;                // wave tile inside block
  const int n0w = (wave & 1) * 64;

  v8f acc[2][4];
#pragma unroll
  for (int i = 0; i < 2; ++i)
#pragma unroll
    for (int j = 0; j < 4; ++j)
#pragma unroll
      for (int e = 0; e < 8; ++e) acc[i][j][e] = 0.0f;

  const int NK = K / 32;
  stage_tiles(A, B, K, m0, n0, 0, aL[0], bL[0], tid);
  int buf = 0;
  for (int kb = 0; kb < NK; ++kb) {
    if (kb + 1 < NK) {
      stage_tiles(A, B, K, m0, n0, (kb + 1) * 32, aL[buf ^ 1], bL[buf ^ 1], tid);
#if HAS_ASYNC
      __builtin_amdgcn_s_wait_asynccnt(6);   // current tile's 6 copies done (in-order)
#endif
    } else {
#if HAS_ASYNC
      __builtin_amdgcn_s_wait_asynccnt(0);
#endif
    }
    __syncthreads();                          // all waves' copies visible

    // A fragment (16-bit 16x32): lane l<16 holds row, K {0..7,16..23};
    // lane l>=16 holds K {8..15,24..31}. B fragment (32x16): lane l<16 holds
    // col, K {0..7,16..23}; lane l>=16 holds K {8..15,24..31}.
    v16bf af[2], bfm[4];
#pragma unroll
    for (int mt = 0; mt < 2; ++mt) {
      const unsigned short* p = &aL[buf][m0w + mt * 16 + l15][lh * 8];
      bf16x8 lo = *(const bf16x8*)p;
      bf16x8 hi = *(const bf16x8*)(p + 16);
      af[mt] = __builtin_shufflevector(lo, hi, 0,1,2,3,4,5,6,7,8,9,10,11,12,13,14,15);
    }
#pragma unroll
    for (int nt = 0; nt < 4; ++nt) {
      const unsigned short* p = &bL[buf][n0w + nt * 16 + l15][lh * 8];
      bf16x8 lo = *(const bf16x8*)p;
      bf16x8 hi = *(const bf16x8*)(p + 16);
      bfm[nt] = __builtin_shufflevector(lo, hi, 0,1,2,3,4,5,6,7,8,9,10,11,12,13,14,15);
    }
#pragma unroll
    for (int mt = 0; mt < 2; ++mt)
#pragma unroll
      for (int nt = 0; nt < 4; ++nt)
        acc[mt][nt] = __builtin_amdgcn_wmma_f32_16x16x32_bf16(
            false, af[mt], false, bfm[nt], (short)0, acc[mt][nt], false, false);

    __syncthreads();                          // buffer safe to overwrite
    buf ^= 1;
  }

  // Epilogue. C layout: lane<16 -> rows +0..7 (vgpr v), lane>=16 -> rows +8..15.
#pragma unroll
  for (int mt = 0; mt < 2; ++mt) {
#pragma unroll
    for (int nt = 0; nt < 4; ++nt) {
      const int col = n0 + n0w + nt * 16 + l15;
      const float bv = bias ? bias[col] : 0.0f;
#pragma unroll
      for (int v = 0; v < 8; ++v) {
        const int rowi = m0 + m0w + mt * 16 + (lh << 3) + v;
        float val = acc[mt][nt][v] + bv;
        if (op == 1) val = 0.5f * val * (1.0f + erff(val * 0.70710678118654752f));
        const size_t idx = (size_t)rowi * N + col;
        if (res)  val += res[idx];
        if (outf) outf[idx] = val;
        if (outb) outb[idx] = f2bf(val);
      }
    }
  }
}

// ---------------------------------------------------------------------------
// Windowed attention: one wave per (head, query). Online softmax over the
// <=129-key window; each lane owns two head-dim components (float2).
// qkv layout: [s, {q,k,v}, h, d] -> s*3072 + which*1024 + h*64 + d.
// ---------------------------------------------------------------------------
__global__ void attn_kernel(const float* __restrict__ qkv,
                            unsigned short* __restrict__ attnb) {
  const int w    = (blockIdx.x * blockDim.x + threadIdx.x) >> 5;
  const int lane = threadIdx.x & 31;
  const int h = w >> 11;            // 2048 queries per head
  const int q = w & (S_LEN - 1);
  const float2 qv = *(const float2*)(qkv + (size_t)q * 3072 + h * HDIM + 2 * lane);
  const int jlo = (q - WHALF > 0) ? (q - WHALF) : 0;
  const int jhi = (q + WHALF < S_LEN - 1) ? (q + WHALF) : (S_LEN - 1);
  const float scale = 0.125f;       // 1/sqrt(64)

  float m = -3.0e38f, l = 0.0f;
  float2 acc; acc.x = 0.0f; acc.y = 0.0f;
  for (int j = jlo; j <= jhi; ++j) {
    const float* kp = qkv + (size_t)j * 3072 + DMODEL + h * HDIM + 2 * lane;
    const float2 kv = *(const float2*)kp;
    float p = qv.x * kv.x + qv.y * kv.y;
#pragma unroll
    for (int off = 16; off > 0; off >>= 1) p += __shfl_xor(p, off, 32);
    const float s = p * scale;
    const float mn = fmaxf(m, s);
    const float c = __expf(m - mn);
    const float e = __expf(s - mn);
    const float* vp = qkv + (size_t)j * 3072 + 2 * DMODEL + h * HDIM + 2 * lane;
    const float2 vv = *(const float2*)vp;
    l = l * c + e;
    acc.x = acc.x * c + e * vv.x;
    acc.y = acc.y * c + e * vv.y;
    m = mn;
  }
  const float inv = 1.0f / l;
  const size_t o = (size_t)q * DMODEL + h * HDIM + 2 * lane;
  attnb[o]     = f2bf(acc.x * inv);
  attnb[o + 1] = f2bf(acc.y * inv);
}

// ---------------------------------------------------------------------------
// Host-side orchestration
// ---------------------------------------------------------------------------
extern "C" void kernel_launch(void* const* d_in, const int* in_sizes, int n_in,
                              void* d_out, int out_size, void* d_ws, size_t ws_size,
                              hipStream_t stream) {
  const float* x     = (const float*)d_in[0];
  const float* qkv_w = (const float*)d_in[1];
  const float* out_w = (const float*)d_in[2];
  const float* ln_g  = (const float*)d_in[3];
  const float* ln_b  = (const float*)d_in[4];
  const float* w1    = (const float*)d_in[5];
  const float* b1    = (const float*)d_in[6];
  const float* w2    = (const float*)d_in[7];
  const float* b2    = (const float*)d_in[8];
  float* out = (float*)d_out;

  char* ws = (char*)d_ws;
  unsigned short* xnb   = (unsigned short*)ws; ws += (size_t)S_LEN * DMODEL * 2;      // 4 MB
  unsigned short* qkvwb = (unsigned short*)ws; ws += (size_t)3 * DMODEL * DMODEL * 2; // 6 MB
  unsigned short* outwb = (unsigned short*)ws; ws += (size_t)DMODEL * DMODEL * 2;     // 2 MB
  unsigned short* w1b   = (unsigned short*)ws; ws += (size_t)4 * DMODEL * DMODEL * 2; // 8 MB
  unsigned short* w2b   = (unsigned short*)ws; ws += (size_t)4 * DMODEL * DMODEL * 2; // 8 MB
  float*          qkvf  = (float*)ws;          ws += (size_t)S_LEN * 3 * DMODEL * 4;  // 24 MB
  unsigned short* attnb = (unsigned short*)ws; ws += (size_t)S_LEN * DMODEL * 2;      // 4 MB
  float*          outf  = (float*)ws;          ws += (size_t)S_LEN * DMODEL * 4;      // 8 MB
  unsigned short* outb  = (unsigned short*)ws; ws += (size_t)S_LEN * DMODEL * 2;      // 4 MB
  unsigned short* hb    = (unsigned short*)ws; ws += (size_t)S_LEN * 4 * DMODEL * 2;  // 16 MB

  // Stage weights in bf16
  cvt_bf16_kernel<<<dim3(1024), 256, 0, stream>>>(qkv_w, qkvwb, 3 * DMODEL * DMODEL);
  cvt_bf16_kernel<<<dim3(1024), 256, 0, stream>>>(out_w, outwb, DMODEL * DMODEL);
  cvt_bf16_kernel<<<dim3(1024), 256, 0, stream>>>(w1, w1b, 4 * DMODEL * DMODEL);
  cvt_bf16_kernel<<<dim3(1024), 256, 0, stream>>>(w2, w2b, 4 * DMODEL * DMODEL);

  // Pre-norm
  layernorm_kernel<<<dim3(S_LEN), 256, 0, stream>>>(x, ln_g, ln_b, xnb);

  // QKV projection: [2048,1024] x [3072,1024]^T -> fp32 qkv
  gemm_bf16_kernel<<<dim3(3 * DMODEL / 128, S_LEN / 64), 128, 0, stream>>>(
      xnb, qkvwb, 3 * DMODEL, DMODEL, nullptr, nullptr, qkvf, nullptr, 0);

  // Windowed attention -> bf16 [2048,1024]
  attn_kernel<<<dim3((NHEAD * S_LEN) / 4), 128, 0, stream>>>(qkvf, attnb);

  // Out projection + residual(x): fp32 outf + bf16 outb
  gemm_bf16_kernel<<<dim3(DMODEL / 128, S_LEN / 64), 128, 0, stream>>>(
      attnb, outwb, DMODEL, DMODEL, nullptr, x, outf, outb, 0);

  // FFN up + exact GELU -> bf16 h
  gemm_bf16_kernel<<<dim3(4 * DMODEL / 128, S_LEN / 64), 128, 0, stream>>>(
      outb, w1b, 4 * DMODEL, DMODEL, b1, nullptr, nullptr, hb, 1);

  // FFN down + bias + residual(outf) -> final fp32 output
  gemm_bf16_kernel<<<dim3(DMODEL / 128, S_LEN / 64), 128, 0, stream>>>(
      hb, w2b, DMODEL, 4 * DMODEL, b2, outf, out, nullptr, 0);
}
